// Head_58394375356500
// MI455X (gfx1250) — compile-verified, matrix operand order checked by reading, not run on hardware
//
#include <hip/hip_runtime.h>

// ---------------------------------------------------------------------------
// Single-head causal attention, B=4 T=4096 C=1024 H=64, fp32 in/out.
// bf16 WMMA (v_wmma_f32_16x16x32_bf16) + flash-attention online softmax.
// ---------------------------------------------------------------------------

typedef __attribute__((ext_vector_type(8)))  float   f32x8;
typedef __attribute__((ext_vector_type(16))) __bf16  bf16x16;
typedef __attribute__((ext_vector_type(8)))  __bf16  bf16x8;

#define NB    4
#define SEQ   4096
#define CDIM  1024
#define HDIM  64

#define WMMA_BF16(a, b, c)                                                \
  __builtin_amdgcn_wmma_f32_16x16x32_bf16(false, (a), false, (b),         \
                                          (short)0, (c), false, false)

// Load a 16-lane 16-bit A/B operand chunk (K=32) given the per-lane K-split:
// lane<16 takes K in {0..7} U {16..23}; lane>=16 takes {8..15} U {24..31}.
static __device__ __forceinline__ bf16x16 ld_ab16(const __bf16* row, int half) {
  bf16x8 lo = *(const bf16x8*)(row + half * 8);
  bf16x8 hi = *(const bf16x8*)(row + 16 + half * 8);
  bf16x16 r;
#pragma unroll
  for (int i = 0; i < 8; ++i) { r[i] = lo[i]; r[i + 8] = hi[i]; }
  return r;
}

static __device__ __forceinline__ float rowmax16(float v) {
#pragma unroll
  for (int m = 1; m < 16; m <<= 1) v = fmaxf(v, __shfl_xor(v, m, 32));
  return v;
}
static __device__ __forceinline__ float rowsum16(float v) {
#pragma unroll
  for (int m = 1; m < 16; m <<= 1) v += __shfl_xor(v, m, 32);
  return v;
}

// ---------------------------------------------------------------------------
// Kernel 0: transpose fp32 [C,H] weights into bf16 [H,C] (coalesced writes).
// ---------------------------------------------------------------------------
__global__ void wt_transpose_kernel(const float* __restrict__ Wq,
                                    const float* __restrict__ Wk,
                                    const float* __restrict__ Wv,
                                    __bf16* __restrict__ WT) {
  int idx = blockIdx.x * blockDim.x + threadIdx.x;   // over 3*H*C
  if (idx >= 3 * HDIM * CDIM) return;
  int m   = idx / (HDIM * CDIM);
  int rem = idx - m * (HDIM * CDIM);
  int h   = rem / CDIM;
  int c   = rem - h * CDIM;
  const float* W = (m == 0) ? Wq : ((m == 1) ? Wk : Wv);
  WT[idx] = (__bf16)W[c * HDIM + h];
}

// ---------------------------------------------------------------------------
// Kernel 1: QKV projection, single pass over x.  One wave = one 16-row tile
// of x (over B*T); holds 12 f32x8 accumulators (q,k,v x 4 N-tiles).
// q (scaled by C^-0.5) and k stored bf16 [B*T][H]; v stored transposed
// bf16 [B][H][T] so the later P@V B-operand loads are contiguous, and so
// the store here packs 8 consecutive rows into one 16-byte b128 store.
// ---------------------------------------------------------------------------
__global__ __launch_bounds__(256)
void qkv_proj_kernel(const float* __restrict__ x,
                     const __bf16* __restrict__ WqT,
                     const __bf16* __restrict__ WkT,
                     const __bf16* __restrict__ WvT,
                     __bf16* __restrict__ qb,
                     __bf16* __restrict__ kb,
                     __bf16* __restrict__ vT) {
  const int lane = threadIdx.x & 31;
  const int wv   = threadIdx.x >> 5;
  const int half = lane >> 4;
  const int ln   = lane & 15;
  const int tileRow = (blockIdx.x * 8 + wv) * 16;   // flat row in [0, B*T)
  const int b       = tileRow / SEQ;
  const float scale = 0.03125f;                      // 1024^-0.5

  const float* xrow = x + (size_t)(tileRow + ln) * CDIM;  // A-operand row

  f32x8 aq[4], ak[4], av[4];
#pragma unroll
  for (int t = 0; t < 4; ++t)
#pragma unroll
    for (int i = 0; i < 8; ++i) { aq[t][i] = 0.0f; ak[t][i] = 0.0f; av[t][i] = 0.0f; }

#pragma unroll 1
  for (int kc = 0; kc < CDIM; kc += 32) {
    // A tile: fp32 x -> bf16, per-lane K-split layout (loaded ONCE per chunk)
    f32x8 f0 = *(const f32x8*)(xrow + kc + half * 8);
    f32x8 f1 = *(const f32x8*)(xrow + kc + 16 + half * 8);
    bf16x16 a;
#pragma unroll
    for (int i = 0; i < 8; ++i) { a[i] = (__bf16)f0[i]; a[i + 8] = (__bf16)f1[i]; }
#pragma unroll
    for (int t = 0; t < 4; ++t) {
      size_t wo = (size_t)(t * 16 + ln) * CDIM + kc;   // column of W = row of WT
      aq[t] = WMMA_BF16(a, ld_ab16(WqT + wo, half), aq[t]);
      ak[t] = WMMA_BF16(a, ld_ab16(WkT + wo, half), ak[t]);
      av[t] = WMMA_BF16(a, ld_ab16(WvT + wo, half), av[t]);
    }
  }

  // Stores per C/D layout: lane holds column h = t*16+ln; VGPR v = row v+half*8.
  const int rloc = tileRow - b * SEQ + half * 8;       // row-in-batch base
#pragma unroll
  for (int t = 0; t < 4; ++t) {
    const int h = t * 16 + ln;
#pragma unroll
    for (int vvi = 0; vvi < 8; ++vvi) {
      size_t fr = (size_t)tileRow + vvi + half * 8;    // flat row
      qb[fr * HDIM + h] = (__bf16)(aq[t][vvi] * scale);
      kb[fr * HDIM + h] = (__bf16)ak[t][vvi];
    }
    // vT: the 8 rows this lane holds are CONTIGUOUS along T -> one b128 store
    bf16x8 pk;
#pragma unroll
    for (int vvi = 0; vvi < 8; ++vvi) pk[vvi] = (__bf16)av[t][vvi];
    *(bf16x8*)(vT + ((size_t)b * HDIM + h) * SEQ + rloc) = pk;
  }
}

// ---------------------------------------------------------------------------
// Kernel 2: causal flash attention.  Block = 128 query rows (8 waves x 16).
// KV tile = 32 keys (two 16-wide S accumulators, one K=32 WMMA for P@V).
// ---------------------------------------------------------------------------
__global__ __launch_bounds__(256)
void flash_attn_kernel(const __bf16* __restrict__ qb,
                       const __bf16* __restrict__ kb,
                       const __bf16* __restrict__ vT,
                       float* __restrict__ out) {
  __shared__ __align__(16) __bf16 pbuf[8][16 * 32];   // per-wave P tile

  const int lane = threadIdx.x & 31;
  const int wv   = threadIdx.x >> 5;
  const int half = lane >> 4;
  const int ln   = lane & 15;
  const int b    = blockIdx.x / (SEQ / 128);
  const int qblk = blockIdx.x % (SEQ / 128);
  const int qlo  = qblk * 128 + wv * 16;              // row-in-batch of wave tile

  // Q as A-operand, both H-chunks, held in registers for the whole loop.
  const __bf16* qrow = qb + (size_t)(b * SEQ + qlo + ln) * HDIM;
  bf16x16 aq[2];
#pragma unroll
  for (int c = 0; c < 2; ++c) aq[c] = ld_ab16(qrow + c * 32, half);

  f32x8 o[4];
#pragma unroll
  for (int t = 0; t < 4; ++t)
#pragma unroll
    for (int i = 0; i < 8; ++i) o[t][i] = 0.0f;
  float mrow[8], lrow[8];
#pragma unroll
  for (int i = 0; i < 8; ++i) { mrow[i] = -__builtin_inff(); lrow[i] = 0.0f; }

  __bf16* pb = pbuf[wv];

#pragma unroll 1
  for (int kv = 0; kv < qlo + 16; kv += 32) {
    // ---- S = Q @ K^T (scale already folded into Q) ----
    f32x8 s0, s1;
#pragma unroll
    for (int i = 0; i < 8; ++i) { s0[i] = 0.0f; s1[i] = 0.0f; }
#pragma unroll
    for (int c = 0; c < 2; ++c) {
      const __bf16* kr0 = kb + (size_t)(b * SEQ + kv + ln) * HDIM + c * 32;
      bf16x16 b0 = ld_ab16(kr0, half);
      bf16x16 b1 = ld_ab16(kr0 + 16 * HDIM, half);
      s0 = WMMA_BF16(aq[c], b0, s0);
      s1 = WMMA_BF16(aq[c], b1, s1);
    }
    if (kv + 32 < qlo + 16)  // gfx1250 global_prefetch of next K tile
      __builtin_prefetch(kb + (size_t)(b * SEQ + kv + 32 + ln) * HDIM, 0, 1);

    // ---- causal mask on the diagonal tile ----
    if (kv + 31 > qlo) {
#pragma unroll
      for (int vvi = 0; vvi < 8; ++vvi) {
        int r = qlo + vvi + half * 8;
        if (kv + ln > r)      s0[vvi] = -__builtin_inff();
        if (kv + 16 + ln > r) s1[vvi] = -__builtin_inff();
      }
    }

    // ---- online softmax update (row stats live per-lane, replicated) ----
    float alpha[8];
#pragma unroll
    for (int vvi = 0; vvi < 8; ++vvi) {
      float smax = rowmax16(fmaxf(s0[vvi], s1[vvi]));
      float mnew = fmaxf(mrow[vvi], smax);
      float p0 = __expf(s0[vvi] - mnew);
      float p1 = __expf(s1[vvi] - mnew);
      alpha[vvi] = __expf(mrow[vvi] - mnew);
      lrow[vvi]  = lrow[vvi] * alpha[vvi] + rowsum16(p0 + p1);
      mrow[vvi]  = mnew;
      // stash P (bf16) into per-wave LDS, row-major 16x32
      int r = vvi + half * 8;
      pb[r * 32 + ln]      = (__bf16)p0;
      pb[r * 32 + 16 + ln] = (__bf16)p1;
    }
#pragma unroll
    for (int t = 0; t < 4; ++t)
#pragma unroll
      for (int vvi = 0; vvi < 8; ++vvi) o[t][vvi] *= alpha[vvi];

    // ---- O += P @ V : re-read P in A layout, V^T rows are contiguous ----
    bf16x16 ap = ld_ab16(pb + ln * 32, half);
#pragma unroll
    for (int t = 0; t < 4; ++t) {
      const __bf16* vr = vT + ((size_t)b * HDIM + t * 16 + ln) * SEQ + kv;
      bf16x16 bv = ld_ab16(vr, half);
      o[t] = WMMA_BF16(ap, bv, o[t]);
    }
  }

  // ---- epilogue: normalize and store fp32 [B][T][H] ----
#pragma unroll
  for (int t = 0; t < 4; ++t) {
    int h = t * 16 + ln;
#pragma unroll
    for (int vvi = 0; vvi < 8; ++vvi) {
      int r = qlo + vvi + half * 8;
      out[((size_t)b * SEQ + r) * HDIM + h] = o[t][vvi] / lrow[vvi];
    }
  }
}

// ---------------------------------------------------------------------------
extern "C" void kernel_launch(void* const* d_in, const int* in_sizes, int n_in,
                              void* d_out, int out_size, void* d_ws, size_t ws_size,
                              hipStream_t stream) {
  const float* x  = (const float*)d_in[0];
  const float* Wq = (const float*)d_in[1];
  const float* Wk = (const float*)d_in[2];
  const float* Wv = (const float*)d_in[3];
  float* out = (float*)d_out;

  char* ws = (char*)d_ws;
  __bf16* WT = (__bf16*)ws;                               // 3 * H * C bf16
  size_t off = (size_t)3 * HDIM * CDIM * sizeof(__bf16);  // 384 KB, 256B-aligned
  __bf16* qb = (__bf16*)(ws + off);                       // B*T*H bf16 (2 MB)
  __bf16* kb = qb + (size_t)NB * SEQ * HDIM;
  __bf16* vT = kb + (size_t)NB * SEQ * HDIM;

  wt_transpose_kernel<<<(3 * HDIM * CDIM + 255) / 256, 256, 0, stream>>>(
      Wq, Wk, Wv, WT);
  qkv_proj_kernel<<<(NB * SEQ) / 128, 256, 0, stream>>>(
      x, WT, WT + HDIM * CDIM, WT + 2 * HDIM * CDIM, qb, kb, vT);
  flash_attn_kernel<<<NB * (SEQ / 128), 256, 0, stream>>>(qb, kb, vT, out);
}
